// DiT_MoMBlock_10179072491668
// MI455X (gfx1250) — matrix operand order repaired and checked by reading, not compiled, not run on hardware
//
#include <hip/hip_runtime.h>

// ---------------------------------------------------------------------------
// DiT MoM block for MI455X (gfx1250): all GEMMs via v_wmma_f32_16x16x32_bf16
// Block tile 128x128 (8 waves as 4m x 2n), wave tile 32x64 (2x4 WMMA tiles).
// ---------------------------------------------------------------------------

typedef __attribute__((ext_vector_type(16))) __bf16 v16bf;
typedef __attribute__((ext_vector_type(8)))  __bf16 v8bf;
typedef __attribute__((ext_vector_type(8)))  float  v8f;

// Problem constants
#define BB     16
#define NN     256
#define DD     1152
#define EE     270
#define TK     27
#define HH     256
#define MLPH   4608
#define SIXD   (6*DD)
#define TOK    (BB*NN)   // 4096

#define LOG2E  1.4426950408889634f

// Branch-free sigmoid: 1/(1+exp(-x)) = rcp(1+exp2(-x*log2e)); v_exp + v_rcp only.
__device__ __forceinline__ float fast_sigmoid(float x) {
    float e = __builtin_amdgcn_exp2f(-x * LOG2E);
    return __builtin_amdgcn_rcpf(1.0f + e);
}

// gelu_tanh(x) = 0.5x(1+tanh(y)) = x * sigmoid(2y),  y = 0.79788456(x+0.044715x^3)
__device__ __forceinline__ float gelu_tanh_f(float x) {
    float y = 0.79788456080286535588f * (x + 0.044715f * x * x * x);
    return x * fast_sigmoid(2.0f * y);
}

__device__ __forceinline__ v8f wmma_bf16(v16bf a, v16bf b, v8f c) {
    // D = A(16x32) * B(32x16) + C, f32 accumulate
    return __builtin_amdgcn_wmma_f32_16x16x32_bf16(false, a, false, b, (short)0, c, false, false);
}

__device__ __forceinline__ v16bf combine16(v8bf lo, v8bf hi) {
    v16bf r;
    #pragma unroll
    for (int i = 0; i < 8; i++) { r[i] = lo[i]; r[i + 8] = hi[i]; }
    return r;
}

// Wave-level 32x64 output tile over K; acc[2][4] of 16x16 f32 tiles.
// A-frag (16x32): lane = row m (lane%16); 16 halves at K[16*hi..+15].
// B-frag (32x16) from Bt (N x K row-major): lane = col n; 16 halves at K[16*hi..+15].
// Row base pointers hoisted so inner loads are base + immediate offsets.
__device__ __forceinline__ void gemm_core24(const __bf16* A, int lda, const __bf16* Bt, int ldb,
                                            int K, int m0, int n0, int lm, int hi, v8f acc[2][4]) {
    const __bf16* ar0 = A + (size_t)(m0 + lm) * lda + 16 * hi;
    const __bf16* ar1 = ar0 + (size_t)16 * lda;
    const __bf16* br[4];
    #pragma unroll
    for (int j = 0; j < 4; j++) br[j] = Bt + (size_t)(n0 + j * 16 + lm) * ldb + 16 * hi;
    #pragma unroll 2
    for (int k0 = 0; k0 < K; k0 += 32) {
        v16bf a0 = combine16(*(const v8bf*)(ar0 + k0), *(const v8bf*)(ar0 + k0 + 8));
        v16bf a1 = combine16(*(const v8bf*)(ar1 + k0), *(const v8bf*)(ar1 + k0 + 8));
        v16bf b[4];
        #pragma unroll
        for (int j = 0; j < 4; j++) b[j] = *(const v16bf*)(br[j] + k0);
        #pragma unroll
        for (int j = 0; j < 4; j++) {
            acc[0][j] = wmma_bf16(a0, b[j], acc[0][j]);
            acc[1][j] = wmma_bf16(a1, b[j], acc[1][j]);
        }
    }
}

// ---------------------------------------------------------------------------
// Elementwise / reduction glue kernels
// ---------------------------------------------------------------------------

__global__ void k_silu(const float* __restrict__ c, float* __restrict__ out, int n) {
    int i = blockIdx.x * blockDim.x + threadIdx.x;
    if (i < n) { float v = c[i]; out[i] = v * fast_sigmoid(v); }
}

// mod[b][j] = adaln_b[j] + dot(silu(c[b]), adaln_w[j])
__global__ void k_adaln(const float* __restrict__ sc, const float* __restrict__ w,
                        const float* __restrict__ bias, float* __restrict__ mod) {
    int idx = blockIdx.x * blockDim.x + threadIdx.x;
    if (idx >= BB * SIXD) return;
    int b = idx / SIXD, j = idx % SIXD;
    const float* a  = sc + b * DD;
    const float* wr = w + (size_t)j * DD;
    float s = bias[j];
    #pragma unroll 4
    for (int k = 0; k < DD; k++) s += a[k] * wr[k];
    mod[idx] = s;
}

// Per-(b,d) stats over N tokens of modx = x*(1+sc_mom)+sh_mom
__global__ void k_colstats(const float* __restrict__ x, const float* __restrict__ mod,
                           float* __restrict__ mu, float* __restrict__ rstd) {
    int idx = blockIdx.x * blockDim.x + threadIdx.x;
    if (idx >= BB * DD) return;
    int b = idx / DD, d = idx % DD;
    float sh = mod[b * SIXD + d];
    float sc = mod[b * SIXD + DD + d];
    const float* xp = x + ((size_t)b * NN) * DD + d;
    float s = 0.f, ss = 0.f;
    for (int n = 0; n < NN; n++) {
        float v = xp[(size_t)n * DD];
        v = v * (1.f + sc) + sh;
        s += v; ss += v * v;
    }
    float m = s * (1.f / NN);
    float var = ss * (1.f / NN) - m * m;
    mu[idx] = m;
    rstd[idx] = rsqrtf(var + 1e-5f);
}

// zT[b][d][n] = ((x*(1+sc)+sh) - mu)*rstd as bf16, via LDS 32x32 transpose
__global__ void k_zt(const float* __restrict__ x, const float* __restrict__ mod,
                     const float* __restrict__ mu, const float* __restrict__ rstd,
                     __bf16* __restrict__ zT) {
    __shared__ float tile[32][33];
    int b = blockIdx.z, d0 = blockIdx.y * 32, n0 = blockIdx.x * 32;
    int tx = threadIdx.x, ty = threadIdx.y;
    int d = d0 + tx;
    float sh = mod[b * SIXD + d];
    float sc = mod[b * SIXD + DD + d];
    float m  = mu[b * DD + d];
    float r  = rstd[b * DD + d];
    for (int i = ty; i < 32; i += 8) {
        float v = x[((size_t)(b * NN + n0 + i)) * DD + d];
        tile[i][tx] = ((v * (1.f + sc) + sh) - m) * r;   // value at (n=n0+i, d=d0+tx)
    }
    __syncthreads();
    for (int i = ty; i < 32; i += 8) {
        zT[((size_t)(b * DD + d0 + i)) * NN + n0 + tx] = (__bf16)tile[tx][i];
    }
}

// Router: logits = mu @ router_w^T, softmax, top-27, normalized weights
__global__ void k_router(const float* __restrict__ mu, const float* __restrict__ rw,
                         float* __restrict__ probs, int* __restrict__ tki,
                         float* __restrict__ tkw) {
    __shared__ float smu[DD];
    __shared__ float sl[EE];
    __shared__ float red[2];
    int b = blockIdx.x, t = threadIdx.x;
    for (int i = t; i < DD; i += 288) smu[i] = mu[b * DD + i];
    __syncthreads();
    if (t < EE) {
        const float* wr = rw + (size_t)t * DD;
        float s = 0.f;
        #pragma unroll 4
        for (int k = 0; k < DD; k++) s += smu[k] * wr[k];
        sl[t] = s;
    }
    __syncthreads();
    if (t == 0) { float mx = -1e30f; for (int e = 0; e < EE; e++) mx = fmaxf(mx, sl[e]); red[0] = mx; }
    __syncthreads();
    if (t < EE) sl[t] = __builtin_amdgcn_exp2f((sl[t] - red[0]) * LOG2E);
    __syncthreads();
    if (t == 0) { float s = 0.f; for (int e = 0; e < EE; e++) s += sl[e]; red[1] = 1.f / s; }
    __syncthreads();
    if (t < EE) { sl[t] *= red[1]; probs[b * EE + t] = sl[t]; }
    __syncthreads();
    if (t == 0) {
        float wsum = 0.f;
        for (int k = 0; k < TK; k++) {
            float best = -1.f; int bi = 0;
            for (int e = 0; e < EE; e++) if (sl[e] > best) { best = sl[e]; bi = e; }
            tki[b * TK + k] = bi; tkw[b * TK + k] = best; wsum += best; sl[bi] = -1.f;
        }
        float inv = 1.f / wsum;
        for (int k = 0; k < TK; k++) tkw[b * TK + k] *= inv;
    }
}

__global__ void k_aux(const float* __restrict__ probs, const int* __restrict__ tki,
                      float* __restrict__ out_aux) {
    __shared__ float s[288];
    int t = threadIdx.x;
    float v = 0.f;
    if (t < EE) {
        float mp = 0.f; int cnt = 0;
        for (int b = 0; b < BB; b++) {
            mp += probs[b * EE + t];
            if (tki[b * TK] == t) cnt++;
        }
        v = (float)EE * (mp / (float)BB) * ((float)cnt / (float)BB);
    }
    s[t] = v; __syncthreads();
    if (t == 0) { float a = 0.f; for (int i = 0; i < 288; i++) a += s[i]; *out_aux = a; }
}

__global__ void k_cvt(const float* __restrict__ src, __bf16* __restrict__ dst, int n) {
    int i = blockIdx.x * blockDim.x + threadIdx.x;
    if (i < n) dst[i] = (__bf16)src[i];
}

// rowbias[b][n] = sum_k tkw[b,k] * fc2_b[e_k][n]
__global__ void k_rowbias(const float* __restrict__ fc2_b, const int* __restrict__ tki,
                          const float* __restrict__ tkw, float* __restrict__ rowbias) {
    int idx = blockIdx.x * blockDim.x + threadIdx.x;
    if (idx >= BB * NN) return;
    int b = idx >> 8, n = idx & 255;
    float s = 0.f;
    for (int k = 0; k < TK; k++)
        s += tkw[b * TK + k] * fc2_b[(size_t)tki[b * TK + k] * NN + n];
    rowbias[idx] = s;
}

// LayerNorm over D per token, then modulate -> bf16 input for MLP GEMM
__global__ void k_ln2(const float* __restrict__ x1, const float* __restrict__ mod,
                      __bf16* __restrict__ mbf) {
    int tkn = blockIdx.x;            // 0..4095
    int b = tkn >> 8;
    int tid = threadIdx.x;
    const float* xr = x1 + (size_t)tkn * DD;
    __shared__ float s1[256], s2[256];
    float s = 0.f, ss = 0.f;
    for (int i = tid; i < DD; i += 256) { float v = xr[i]; s += v; ss += v * v; }
    s1[tid] = s; s2[tid] = ss; __syncthreads();
    for (int o = 128; o > 0; o >>= 1) {
        if (tid < o) { s1[tid] += s1[tid + o]; s2[tid] += s2[tid + o]; }
        __syncthreads();
    }
    float m = s1[0] * (1.f / DD);
    float var = s2[0] * (1.f / DD) - m * m;
    float r = rsqrtf(var + 1e-6f);
    const float* md = mod + b * SIXD;
    for (int i = tid; i < DD; i += 256) {
        float v = (xr[i] - m) * r;
        v = v * (1.f + md[4 * DD + i]) + md[3 * DD + i];   // sc_mlp, sh_mlp
        mbf[(size_t)tkn * DD + i] = (__bf16)v;
    }
}

// ---------------------------------------------------------------------------
// WMMA GEMM kernels (block 256 thr = 8 waves as 4m x 2n; block tile 128x128)
// ---------------------------------------------------------------------------

// fc1 (batch group): hT[lb,k] (D x H) = tkw * gelu(zT[b] @ w1[e]^T + fc1_b[e]) -> bf16
__global__ void k_gemm_fc1(const __bf16* __restrict__ zT, const __bf16* __restrict__ w1bf,
                           const float* __restrict__ fc1_b, const int* __restrict__ tki,
                           const float* __restrict__ tkw, int gb, __bf16* __restrict__ hT) {
    int k  = blockIdx.y;
    int lb = blockIdx.z;
    int b  = gb + lb;
    int e  = tki[b * TK + k];
    float wk = tkw[b * TK + k];
    const __bf16* A  = zT + (size_t)b * DD * NN;              // D x N, lda = NN
    const __bf16* Bt = w1bf + (size_t)e * HH * NN;            // H x N, ldb = NN
    const float* bias = fc1_b + (size_t)e * HH;
    __bf16* out = hT + ((size_t)lb * TK + k) * DD * HH;

    int bm = blockIdx.x % (DD / 128);   // 9
    int bn = blockIdx.x / (DD / 128);   // 2
    int lane = threadIdx.x & 31, wave = threadIdx.x >> 5;
    int lm = lane & 15, hi = lane >> 4;
    int m0 = bm * 128 + (wave >> 1) * 32;
    int n0 = bn * 128 + (wave & 1) * 64;

    v8f acc[2][4] = {};
    gemm_core24(A, NN, Bt, NN, NN, m0, n0, lm, hi, acc);

    #pragma unroll
    for (int i = 0; i < 2; i++)
        #pragma unroll
        for (int j = 0; j < 4; j++) {
            int col = n0 + j * 16 + lm;
            float bv = bias[col];
            #pragma unroll
            for (int r = 0; r < 8; r++) {
                int row = m0 + i * 16 + hi * 8 + r;
                out[(size_t)row * HH + col] = (__bf16)(wk * gelu_tanh_f(acc[i][j][r] + bv));
            }
        }
}

// fc2 (batch group): mom[b] (N x D) = sum over continuous K=27*256 of
//   w2[e_k] @ hT[lb,k]^T   (+ precomputed rowbias), -> bf16
__global__ void k_gemm_fc2(const __bf16* __restrict__ w2bf, const __bf16* __restrict__ hT,
                           const float* __restrict__ rowbias, const int* __restrict__ tki,
                           int gb, __bf16* __restrict__ mombf) {
    int lb = blockIdx.x / 18;
    int t  = blockIdx.x % 18;
    int b  = gb + lb;
    int bm = t % (NN / 128);            // 2
    int bn = t / (NN / 128);            // 9
    int lane = threadIdx.x & 31, wave = threadIdx.x >> 5;
    int lm = lane & 15, hi = lane >> 4;
    int m0 = bm * 128 + (wave >> 1) * 32;
    int n0 = bn * 128 + (wave & 1) * 64;

    v8f acc[2][4] = {};
    for (int k = 0; k < TK; k++) {
        int e = tki[b * TK + k];
        const __bf16* A  = w2bf + (size_t)e * NN * HH;           // N x H, lda = HH
        const __bf16* Bt = hT + ((size_t)lb * TK + k) * DD * HH; // D x H, ldb = HH
        gemm_core24(A, HH, Bt, HH, HH, m0, n0, lm, hi, acc);
    }
    const float* rb = rowbias + b * NN;
    __bf16* out = mombf + (size_t)b * NN * DD;
    #pragma unroll
    for (int i = 0; i < 2; i++)
        #pragma unroll
        for (int j = 0; j < 4; j++) {
            int col = n0 + j * 16 + lm;
            #pragma unroll
            for (int r = 0; r < 8; r++) {
                int row = m0 + i * 16 + hi * 8 + r;
                out[(size_t)row * DD + col] = (__bf16)(acc[i][j][r] + rb[row]);
            }
        }
}

// Generic C = A @ Bt^T + bias with epilogues:
//  MODE 0: out_bf16 = gelu(v)                          (mlp fc1)
//  MODE 1: out_f32 = resid + mod[b][2D + col] * v      (out_proj + g_mom residual)
//  MODE 2: out_f32 = resid + mod[b][5D + col] * v      (mlp fc2 + g_mlp residual)
template <int MODE>
__global__ void k_gemm(const __bf16* __restrict__ A, const __bf16* __restrict__ Bt,
                       const float* __restrict__ bias, const float* __restrict__ resid,
                       const float* __restrict__ mod, float* __restrict__ outF,
                       __bf16* __restrict__ outB, int Mtiles, int Nc, int K) {
    int bm = blockIdx.x % Mtiles;
    int bn = blockIdx.x / Mtiles;
    int lane = threadIdx.x & 31, wave = threadIdx.x >> 5;
    int lm = lane & 15, hi = lane >> 4;
    int m0 = bm * 128 + (wave >> 1) * 32;
    int n0 = bn * 128 + (wave & 1) * 64;

    v8f acc[2][4] = {};
    gemm_core24(A, K, Bt, K, K, m0, n0, lm, hi, acc);

    #pragma unroll
    for (int i = 0; i < 2; i++)
        #pragma unroll
        for (int j = 0; j < 4; j++) {
            int col = n0 + j * 16 + lm;
            float bv = bias[col];
            #pragma unroll
            for (int r = 0; r < 8; r++) {
                int row = m0 + i * 16 + hi * 8 + r;
                float v = acc[i][j][r] + bv;
                if (MODE == 0) {
                    outB[(size_t)row * Nc + col] = (__bf16)gelu_tanh_f(v);
                } else {
                    int b = row >> 8;   // 256 tokens per batch
                    float g = mod[b * SIXD + (MODE == 1 ? 2 * DD : 5 * DD) + col];
                    outF[(size_t)row * Nc + col] = resid[(size_t)row * Nc + col] + g * v;
                }
            }
        }
}

// ---------------------------------------------------------------------------
// Host launcher
// ---------------------------------------------------------------------------

extern "C" void kernel_launch(void* const* d_in, const int* in_sizes, int n_in,
                              void* d_out, int out_size, void* d_ws, size_t ws_size,
                              hipStream_t stream) {
    (void)in_sizes; (void)n_in; (void)out_size;
    const float* x         = (const float*)d_in[0];
    const float* c         = (const float*)d_in[1];
    const float* router_w  = (const float*)d_in[2];
    const float* fc1_w     = (const float*)d_in[3];
    const float* fc1_b     = (const float*)d_in[4];
    const float* fc2_w     = (const float*)d_in[5];
    const float* fc2_b     = (const float*)d_in[6];
    const float* out_proj_w= (const float*)d_in[7];
    const float* out_proj_b= (const float*)d_in[8];
    const float* adaln_w   = (const float*)d_in[9];
    const float* adaln_b   = (const float*)d_in[10];
    const float* mlp_w1    = (const float*)d_in[11];
    const float* mlp_b1    = (const float*)d_in[12];
    const float* mlp_w2    = (const float*)d_in[13];
    const float* mlp_b2    = (const float*)d_in[14];
    float* out = (float*)d_out;

    // workspace bump allocator (256B aligned)
    char* wsp = (char*)d_ws;
    size_t off = 0;
    auto alloc = [&](size_t bytes) -> void* {
        void* p = wsp + off;
        off += (bytes + 255) & ~(size_t)255;
        return p;
    };
    float*  siluc  = (float*)alloc((size_t)BB * DD * 4);
    float*  mod    = (float*)alloc((size_t)BB * SIXD * 4);
    float*  mu     = (float*)alloc((size_t)BB * DD * 4);
    float*  rstd   = (float*)alloc((size_t)BB * DD * 4);
    float*  probs  = (float*)alloc((size_t)BB * EE * 4);
    int*    tki    = (int*)  alloc((size_t)BB * TK * 4);
    float*  tkw    = (float*)alloc((size_t)BB * TK * 4);
    float*  rowb   = (float*)alloc((size_t)BB * NN * 4);
    __bf16* zT     = (__bf16*)alloc((size_t)BB * DD * NN * 2);
    __bf16* w1bf   = (__bf16*)alloc((size_t)EE * HH * NN * 2);
    __bf16* w2bf   = (__bf16*)alloc((size_t)EE * NN * HH * 2);
    __bf16* m1bf   = (__bf16*)alloc((size_t)MLPH * DD * 2);
    __bf16* m2bf   = (__bf16*)alloc((size_t)DD * MLPH * 2);
    __bf16* opbf   = (__bf16*)alloc((size_t)DD * DD * 2);
    __bf16* mombf  = (__bf16*)alloc((size_t)TOK * DD * 2);
    float*  x1     = (float*)alloc((size_t)TOK * DD * 4);
    __bf16* mbf    = (__bf16*)alloc((size_t)TOK * DD * 2);
    __bf16* hidbf  = (__bf16*)alloc((size_t)TOK * MLPH * 2);

    // hT staging for GRP batches at a time; pick largest GRP that fits ws_size.
    // Deterministic: depends only on (constant) ws_size.
    int grp = BB;
    while (grp > 1 && off + (size_t)grp * TK * DD * HH * 2 > ws_size) grp >>= 1;
    __bf16* hT = (__bf16*)alloc((size_t)grp * TK * DD * HH * 2);

    // 1) modulation vector
    k_silu<<<(BB * DD + 255) / 256, 256, 0, stream>>>(c, siluc, BB * DD);
    k_adaln<<<(BB * SIXD + 255) / 256, 256, 0, stream>>>(siluc, adaln_w, adaln_b, mod);

    // 2) token-axis stats + normalized transposed activations (bf16)
    k_colstats<<<(BB * DD + 255) / 256, 256, 0, stream>>>(x, mod, mu, rstd);
    k_zt<<<dim3(NN / 32, DD / 32, BB), dim3(32, 8), 0, stream>>>(x, mod, mu, rstd, zT);

    // 3) router + aux loss + fc2 row bias
    k_router<<<BB, 288, 0, stream>>>(mu, router_w, probs, tki, tkw);
    k_aux<<<1, 288, 0, stream>>>(probs, tki, out + (size_t)BB * NN * DD);
    k_rowbias<<<(BB * NN + 255) / 256, 256, 0, stream>>>(fc2_b, tki, tkw, rowb);

    // 4) weight conversions f32 -> bf16
    {
        int n1 = EE * HH * NN, n2 = EE * NN * HH;
        int n3 = MLPH * DD, n4 = DD * MLPH, n5 = DD * DD;
        k_cvt<<<(n1 + 255) / 256, 256, 0, stream>>>(fc1_w, w1bf, n1);
        k_cvt<<<(n2 + 255) / 256, 256, 0, stream>>>(fc2_w, w2bf, n2);
        k_cvt<<<(n3 + 255) / 256, 256, 0, stream>>>(mlp_w1, m1bf, n3);
        k_cvt<<<(n4 + 255) / 256, 256, 0, stream>>>(mlp_w2, m2bf, n4);
        k_cvt<<<(n5 + 255) / 256, 256, 0, stream>>>(out_proj_w, opbf, n5);
    }

    // 5) expert MoM: grouped fc1 (weighted-gelu epilogue) then single-pass fc2
    for (int gb = 0; gb < BB; gb += grp) {
        k_gemm_fc1<<<dim3((DD / 128) * (HH / 128), TK, grp), 256, 0, stream>>>(
            zT, w1bf, fc1_b, tki, tkw, gb, hT);
        k_gemm_fc2<<<18 * grp, 256, 0, stream>>>(
            w2bf, hT, rowb, tki, gb, mombf);
    }

    // 6) out_proj + g_mom residual -> x1
    k_gemm<1><<<(TOK / 128) * (DD / 128), 256, 0, stream>>>(
        mombf, opbf, out_proj_b, x, mod, x1, nullptr, TOK / 128, DD, DD);

    // 7) LayerNorm over D + mlp modulation -> bf16
    k_ln2<<<TOK, 256, 0, stream>>>(x1, mod, mbf);

    // 8) MLP fc1 (gelu, bf16) and fc2 (+ g_mlp residual -> d_out)
    k_gemm<0><<<(TOK / 128) * (MLPH / 128), 256, 0, stream>>>(
        mbf, m1bf, mlp_b1, nullptr, nullptr, nullptr, hidbf, TOK / 128, MLPH, DD);
    k_gemm<2><<<(TOK / 128) * (DD / 128), 256, 0, stream>>>(
        hidbf, m2bf, mlp_b2, x1, mod, out, nullptr, TOK / 128, DD, MLPH);
}
